// ImprovedMultimodalGNN_83150566851218
// MI455X (gfx1250) — matrix operand order repaired and checked by reading, not compiled, not run on hardware
//
#include <hip/hip_runtime.h>
#include <hip/hip_bf16.h>

// ---------------------------------------------------------------------------
// Problem constants (match reference)
// ---------------------------------------------------------------------------
#define NN   102400
#define EE   3276800
#define BB   256
#define NPG  400          // nodes per graph (N/B)
#define EPSV 1e-5f

typedef __attribute__((ext_vector_type(16))) _Float16     v16h;
typedef __attribute__((ext_vector_type(8)))  float        v8f;
typedef __attribute__((ext_vector_type(4)))  unsigned int v4u;

union HF { v4u u[2]; v16h h; };
union HU { _Float16 f; unsigned short s; };

// ---------------------------------------------------------------------------
// Utility kernels
// ---------------------------------------------------------------------------
__global__ void k_zero_f32(float* p, size_t n) {
    size_t i = blockIdx.x * (size_t)blockDim.x + threadIdx.x;
    size_t stride = (size_t)gridDim.x * blockDim.x;
    for (; i < n; i += stride) p[i] = 0.0f;
}

__global__ void k_f32_to_f16(const float* __restrict__ in, _Float16* __restrict__ out, size_t n) {
    size_t i = blockIdx.x * (size_t)blockDim.x + threadIdx.x;
    if (i < n) out[i] = (_Float16)in[i];
}

// ---------------------------------------------------------------------------
// Degree / norm: deg starts at 1 (self loop), +1 per incoming edge, then rsqrt
// ---------------------------------------------------------------------------
__global__ void k_deg_init(float* deg, int n) {
    int i = blockIdx.x * blockDim.x + threadIdx.x;
    if (i < n) deg[i] = 1.0f;
}
__global__ void k_deg_edges(const int* __restrict__ ei, float* deg, int E) {
    int e = blockIdx.x * blockDim.x + threadIdx.x;
    if (e < E) atomicAdd(&deg[ei[E + e]], 1.0f);
}
__global__ void k_deg_rsqrt(float* dinv, int n) {
    int i = blockIdx.x * blockDim.x + threadIdx.x;
    if (i < n) dinv[i] = rsqrtf(dinv[i]);
}

// ---------------------------------------------------------------------------
// Weight pre-pack into CDNA5 WMMA B-fragment layout:
// Wp[((nt*kT + kt)*32 + lane)*8 + v] = pack(W[k][n], W[k+1][n])
//   n = nt*16 + (lane&15), k = kt*32 + (lane>>4)*8 + 2*v + (v>>2)*8
// ---------------------------------------------------------------------------
__global__ void k_pack_w(const float* __restrict__ W, unsigned int* __restrict__ Wp,
                         int K, int Nout) {
    int kT = K >> 5;
    int nT = Nout >> 4;
    int idx = blockIdx.x * blockDim.x + threadIdx.x;
    int total = nT * kT * 32;
    if (idx >= total) return;
    int lane = idx & 31;
    int kt = (idx >> 5) % kT;
    int nt = (idx >> 5) / kT;
    int n = nt * 16 + (lane & 15);
    int kb = kt * 32 + ((lane >> 4) << 3);
    unsigned int* dst = Wp + (size_t)idx * 8;
#pragma unroll
    for (int v = 0; v < 8; ++v) {
        HU lo, hi;
        lo.f = (_Float16)W[(size_t)(kb + 2 * v + ((v >> 2) << 3)) * Nout + n];
        hi.f = (_Float16)W[(size_t)(kb + 2 * v + ((v >> 2) << 3) + 1) * Nout + n];
        dst[v] = (unsigned int)lo.s | ((unsigned int)hi.s << 16);
    }
}

// ---------------------------------------------------------------------------
// WMMA GEMM: C[M x Nout] = A[M x K] (f16 row-major) @ W (pre-packed B frags)
// one wave per 16x16 output tile; K step = 32 via v_wmma_f32_16x16x32_f16
// A-fragment: lane reads two contiguous 16B chunks:
//   k0 = kt*32 + (lane>>4)*8     (VGPRs 0..3)
//   k1 = k0 + 16                 (VGPRs 4..7)
// ---------------------------------------------------------------------------
__global__ __launch_bounds__(256)
void k_gemm_wmma(const _Float16* __restrict__ A, const unsigned int* __restrict__ Wp,
                 float* __restrict__ C, int M, int K, int Nout) {
    int wave = blockIdx.x * (blockDim.x >> 5) + (threadIdx.x >> 5);
    int lane = threadIdx.x & 31;
    int nT = Nout >> 4;
    int mT = M >> 4;
    if (wave >= mT * nT) return;
    int tm = wave / nT;
    int tn = wave % nT;
    int kT = K >> 5;

    int row  = tm * 16 + (lane & 15);
    int koff = (lane >> 4) * 8;
    const _Float16* arow = A + (size_t)row * K + koff;
    const unsigned int* bp = Wp + ((size_t)(tn * kT) * 32 + lane) * 8;

    v8f acc = {};
    for (int kt = 0; kt < kT; ++kt) {
        HF a, b;
        a.u[0] = *(const v4u*)(arow + kt * 32);
        a.u[1] = *(const v4u*)(arow + kt * 32 + 16);
        b.u[0] = *(const v4u*)(bp + kt * 256);
        b.u[1] = *(const v4u*)(bp + kt * 256 + 4);
        acc = __builtin_amdgcn_wmma_f32_16x16x32_f16(
            false, a.h, false, b.h, (short)0, acc, false, false);
    }
    // C layout: VGPR r -> M = tm*16 + r + 8*(lane>>4), N = tn*16 + (lane&15)
    int cn = tn * 16 + (lane & 15);
    int rb = tm * 16 + 8 * (lane >> 4);
#pragma unroll
    for (int r = 0; r < 8; ++r)
        C[(size_t)(rb + r) * Nout + cn] = acc[r];
}

// ---------------------------------------------------------------------------
// Edge aggregation: agg[d] += lin[s] * dinv[s]*dinv[d]  (self loops appended)
// one thread per (edge, 4 features)
// ---------------------------------------------------------------------------
__global__ void k_aggregate(const int* __restrict__ ei, const float* __restrict__ dinv,
                            const float* __restrict__ lin, float* agg,
                            int E, int N, int F) {
    long long t = blockIdx.x * (long long)blockDim.x + threadIdx.x;
    int fch = F >> 2;
    long long total = (long long)(E + N) * fch;
    if (t >= total) return;
    int e  = (int)(t / fch);
    int f4 = ((int)(t % fch)) << 2;
    int s, d;
    if (e < E) { s = ei[e]; d = ei[E + e]; } else { s = d = e - E; }
    float nrm = dinv[s] * dinv[d];
    const float4 v = *(const float4*)(lin + (size_t)s * F + f4);
    float* o = agg + (size_t)d * F + f4;
    atomicAdd(o + 0, v.x * nrm);
    atomicAdd(o + 1, v.y * nrm);
    atomicAdd(o + 2, v.z * nrm);
    atomicAdd(o + 3, v.w * nrm);
}

// ---------------------------------------------------------------------------
// BatchNorm stats over N rows, F cols (F <= 64). stats[0..F)=sum, [F..2F)=sumsq
// (bias cancels in BN, so stats are computed on agg directly)
// ---------------------------------------------------------------------------
__global__ void k_bn_stats(const float* __restrict__ X, float* stats,
                           int Nrows, int F, int rowsPerBlock) {
    __shared__ float ssum[64], ssq[64];
    int tid = threadIdx.x;
    if (tid < F) { ssum[tid] = 0.f; ssq[tid] = 0.f; }
    __syncthreads();
    int col = tid % F;
    int rstride = blockDim.x / F;
    int r0 = blockIdx.x * rowsPerBlock + tid / F;
    int rend = (blockIdx.x + 1) * rowsPerBlock;
    if (rend > Nrows) rend = Nrows;
    float s = 0.f, q = 0.f;
    for (int r = r0; r < rend; r += rstride) {
        float v = X[(size_t)r * F + col];
        s += v; q = fmaf(v, v, q);
    }
    atomicAdd(&ssum[col], s);
    atomicAdd(&ssq[col], q);
    __syncthreads();
    if (tid < F) {
        atomicAdd(&stats[tid], ssum[tid]);
        atomicAdd(&stats[F + tid], ssq[tid]);
    }
}

__global__ void k_bn_apply_relu_f16(const float* __restrict__ X, const float* __restrict__ stats,
                                    const float* __restrict__ g, const float* __restrict__ b,
                                    _Float16* __restrict__ out, int Nrows, int F) {
    size_t i = blockIdx.x * (size_t)blockDim.x + threadIdx.x;
    size_t total = (size_t)Nrows * F;
    if (i >= total) return;
    int col = (int)(i % F);
    float invN = 1.0f / (float)Nrows;
    float mean = stats[col] * invN;
    float var  = stats[F + col] * invN - mean * mean;
    float y = (X[i] - mean) * rsqrtf(var + EPSV) * g[col] + b[col];
    out[i] = (_Float16)fmaxf(y, 0.0f);
}

__global__ void k_bias_relu(const float* __restrict__ X, const float* __restrict__ bias,
                            float* __restrict__ out, size_t n, int F) {
    size_t i = blockIdx.x * (size_t)blockDim.x + threadIdx.x;
    if (i >= n) return;
    out[i] = fmaxf(X[i] + bias[i % F], 0.0f);
}

// ---------------------------------------------------------------------------
// Per-graph mean/max pooling of gf[N,16] into h96[B,96] cols 0..31
// batch is contiguous blocks of NPG nodes per graph.
// ---------------------------------------------------------------------------
__global__ void k_pool(const float* __restrict__ gf, float* __restrict__ h96, int npg) {
    __shared__ float rs[256], rm[256];
    int tid = threadIdx.x;
    int col = tid & 15;
    const float* base = gf + (size_t)blockIdx.x * npg * 16;
    float s = 0.f, m = -3.402823e38f;
    for (int r = (tid >> 4); r < npg; r += 16) {
        float v = base[r * 16 + col];
        s += v;
        if (v > m) m = v;
    }
    rs[tid] = s; rm[tid] = m;
    __syncthreads();
    for (int off = 128; off >= 16; off >>= 1) {
        if (tid < off) {
            rs[tid] += rs[tid + off];
            rm[tid] = fmaxf(rm[tid], rm[tid + off]);
        }
        __syncthreads();
    }
    if (tid < 16) {
        h96[(size_t)blockIdx.x * 96 + tid]      = rs[tid] / (float)npg;
        h96[(size_t)blockIdx.x * 96 + 16 + tid] = rm[tid];
    }
}

// ---------------------------------------------------------------------------
// Small dense: Y[b, offY + j] = (relu?)( X[b,:] @ W + bias )   (row stride ldY)
// ---------------------------------------------------------------------------
__global__ void k_lin(const float* __restrict__ X, const float* __restrict__ W,
                      const float* __restrict__ bias, float* __restrict__ Y,
                      int Bn, int K, int Nout, int ldY, int offY, int doRelu) {
    int idx = blockIdx.x * blockDim.x + threadIdx.x;
    if (idx >= Bn * Nout) return;
    int b = idx / Nout, j = idx % Nout;
    float a = bias[j];
    const float* xr = X + (size_t)b * K;
    for (int k = 0; k < K; ++k) a = fmaf(xr[k], W[(size_t)k * Nout + j], a);
    if (doRelu) a = fmaxf(a, 0.0f);
    Y[(size_t)b * ldY + offY + j] = a;
}

__global__ void k_bn_small_stats(const float* __restrict__ X, float* stats, int Bn, int F) {
    int j = threadIdx.x + blockIdx.x * blockDim.x;
    if (j >= F) return;
    float s = 0.f, q = 0.f;
    for (int b = 0; b < Bn; ++b) {
        float v = X[(size_t)b * F + j];
        s += v; q = fmaf(v, v, q);
    }
    float mean = s / (float)Bn;
    float var  = q / (float)Bn - mean * mean;
    stats[j] = mean;
    stats[F + j] = rsqrtf(var + EPSV);
}

__global__ void k_bn_small_apply_relu(float* X, const float* __restrict__ stats,
                                      const float* __restrict__ g, const float* __restrict__ b,
                                      int Bn, int F) {
    int idx = blockIdx.x * blockDim.x + threadIdx.x;
    if (idx >= Bn * F) return;
    int j = idx % F;
    float y = (X[idx] - stats[j]) * stats[F + j] * g[j] + b[j];
    X[idx] = fmaxf(y, 0.0f);
}

__global__ void k_cls3_logsoftmax(const float* __restrict__ X, const float* __restrict__ W,
                                  const float* __restrict__ bias, float* __restrict__ out, int Bn) {
    int b = blockIdx.x * blockDim.x + threadIdx.x;
    if (b >= Bn) return;
    float o[3];
#pragma unroll
    for (int c = 0; c < 3; ++c) {
        float a = bias[c];
        for (int k = 0; k < 32; ++k) a = fmaf(X[(size_t)b * 32 + k], W[k * 3 + c], a);
        o[c] = a;
    }
    float mx = fmaxf(o[0], fmaxf(o[1], o[2]));
    float se = expf(o[0] - mx) + expf(o[1] - mx) + expf(o[2] - mx);
    float ls = mx + logf(se);
#pragma unroll
    for (int c = 0; c < 3; ++c) out[(size_t)b * 3 + c] = o[c] - ls;
}

// ---------------------------------------------------------------------------
// Host-side orchestration
// ---------------------------------------------------------------------------
static inline unsigned int cdiv(long long a, long long b) { return (unsigned int)((a + b - 1) / b); }

extern "C" void kernel_launch(void* const* d_in, const int* in_sizes, int n_in,
                              void* d_out, int out_size, void* d_ws, size_t ws_size,
                              hipStream_t stream) {
    const float* x    = (const float*)d_in[0];
    const int*   ei   = (const int*)  d_in[1];
    // d_in[2] = batch (contiguous NPG blocks; not needed)
    const float* mri  = (const float*)d_in[3];
    const float* cog  = (const float*)d_in[4];
    const float* clin = (const float*)d_in[5];
    const float* gen  = (const float*)d_in[6];
    const float* W1 = (const float*)d_in[7];  const float* b1 = (const float*)d_in[8];
    const float* W2 = (const float*)d_in[9];  const float* b2 = (const float*)d_in[10];
    const float* W3 = (const float*)d_in[11]; const float* b3 = (const float*)d_in[12];
    const float* bn1_g = (const float*)d_in[13]; const float* bn1_b = (const float*)d_in[14];
    const float* bn2_g = (const float*)d_in[15]; const float* bn2_b = (const float*)d_in[16];
    const float* mriW = (const float*)d_in[17]; const float* mrib = (const float*)d_in[18];
    const float* cogW = (const float*)d_in[19]; const float* cogb = (const float*)d_in[20];
    const float* clinW = (const float*)d_in[21]; const float* clinb = (const float*)d_in[22];
    const float* genW = (const float*)d_in[23]; const float* genb = (const float*)d_in[24];
    const float* c1W = (const float*)d_in[25]; const float* c1b = (const float*)d_in[26];
    const float* cbn1_g = (const float*)d_in[27]; const float* cbn1_b = (const float*)d_in[28];
    const float* c2W = (const float*)d_in[29]; const float* c2b = (const float*)d_in[30];
    const float* cbn2_g = (const float*)d_in[31]; const float* cbn2_b = (const float*)d_in[32];
    const float* c3W = (const float*)d_in[33]; const float* c3b = (const float*)d_in[34];
    float* out = (float*)d_out;

    (void)in_sizes; (void)n_in; (void)out_size; (void)ws_size;

    // ---- workspace carve-up (reused across layers) ----
    char* ws = (char*)d_ws;
    size_t off = 0;
    auto take = [&](size_t bytes) -> void* {
        void* p = ws + off;
        off = (off + bytes + 255) & ~(size_t)255;
        return p;
    };
    _Float16* xh  = (_Float16*)take((size_t)NN * 128 * 2);
    float*    lin = (float*)   take((size_t)NN * 64 * 4);
    float*    agg = (float*)   take((size_t)NN * 64 * 4);
    _Float16* hh  = (_Float16*)take((size_t)NN * 64 * 2);
    float*    gf  = (float*)   take((size_t)NN * 16 * 4);
    float*    dinv= (float*)   take((size_t)NN * 4);
    unsigned* Wp1 = (unsigned*)take(4 * 4 * 32 * 8 * 4);
    unsigned* Wp2 = (unsigned*)take(2 * 2 * 32 * 8 * 4);
    unsigned* Wp3 = (unsigned*)take(1 * 1 * 32 * 8 * 4);
    float*    st  = (float*)   take(128 * 4);
    float*    h96 = (float*)   take((size_t)BB * 96 * 4);
    float*    t1  = (float*)   take((size_t)BB * 64 * 4);
    float*    t2  = (float*)   take((size_t)BB * 32 * 4);
    float*    cst = (float*)   take(128 * 4);

    const int TB = 256;

    // ---- prep: f16 features, packed weights, degree norms ----
    k_f32_to_f16<<<cdiv((long long)NN * 128, TB), TB, 0, stream>>>(x, xh, (size_t)NN * 128);
    k_pack_w<<<cdiv(4 * 4 * 32, TB), TB, 0, stream>>>(W1, Wp1, 128, 64);
    k_pack_w<<<cdiv(2 * 2 * 32, TB), TB, 0, stream>>>(W2, Wp2, 64, 32);
    k_pack_w<<<cdiv(1 * 1 * 32, TB), TB, 0, stream>>>(W3, Wp3, 32, 16);
    k_deg_init<<<cdiv(NN, TB), TB, 0, stream>>>(dinv, NN);
    k_deg_edges<<<cdiv(EE, TB), TB, 0, stream>>>(ei, dinv, EE);
    k_deg_rsqrt<<<cdiv(NN, TB), TB, 0, stream>>>(dinv, NN);

    // ---- GCN layer 1: 128 -> 64 ----
    {
        const int K = 128, F = 64;
        unsigned waves = (NN / 16) * (F / 16);
        k_gemm_wmma<<<cdiv(waves, 8), TB, 0, stream>>>(xh, Wp1, lin, NN, K, F);
        k_zero_f32<<<4096, TB, 0, stream>>>(agg, (size_t)NN * F);
        k_aggregate<<<cdiv((long long)(EE + NN) * (F / 4), TB), TB, 0, stream>>>(
            ei, dinv, lin, agg, EE, NN, F);
        k_zero_f32<<<1, TB, 0, stream>>>(st, 2 * F);
        k_bn_stats<<<cdiv(NN, 1024), TB, 0, stream>>>(agg, st, NN, F, 1024);
        k_bn_apply_relu_f16<<<cdiv((long long)NN * F, TB), TB, 0, stream>>>(
            agg, st, bn1_g, bn1_b, hh, NN, F);
        (void)b1; // bias cancels inside BatchNorm
    }
    // ---- GCN layer 2: 64 -> 32 ----
    {
        const int K = 64, F = 32;
        unsigned waves = (NN / 16) * (F / 16);
        k_gemm_wmma<<<cdiv(waves, 8), TB, 0, stream>>>(hh, Wp2, lin, NN, K, F);
        k_zero_f32<<<4096, TB, 0, stream>>>(agg, (size_t)NN * F);
        k_aggregate<<<cdiv((long long)(EE + NN) * (F / 4), TB), TB, 0, stream>>>(
            ei, dinv, lin, agg, EE, NN, F);
        k_zero_f32<<<1, TB, 0, stream>>>(st, 2 * F);
        k_bn_stats<<<cdiv(NN, 1024), TB, 0, stream>>>(agg, st, NN, F, 1024);
        k_bn_apply_relu_f16<<<cdiv((long long)NN * F, TB), TB, 0, stream>>>(
            agg, st, bn2_g, bn2_b, hh, NN, F);
        (void)b2;
    }
    // ---- GCN layer 3: 32 -> 16, relu(agg + b3), no BN ----
    {
        const int K = 32, F = 16;
        unsigned waves = (NN / 16) * (F / 16);
        k_gemm_wmma<<<cdiv(waves, 8), TB, 0, stream>>>(hh, Wp3, lin, NN, K, F);
        k_zero_f32<<<4096, TB, 0, stream>>>(agg, (size_t)NN * F);
        k_aggregate<<<cdiv((long long)(EE + NN) * (F / 4), TB), TB, 0, stream>>>(
            ei, dinv, lin, agg, EE, NN, F);
        k_bias_relu<<<cdiv((long long)NN * F, TB), TB, 0, stream>>>(
            agg, b3, gf, (size_t)NN * F, F);
    }

    // ---- pooling (mean|max) -> h96[:, 0:32] ----
    k_pool<<<BB, TB, 0, stream>>>(gf, h96, NPG);

    // ---- multimodal encoders -> h96[:, 32:96] ----
    k_lin<<<cdiv(BB * 16, TB), TB, 0, stream>>>(mri,  mriW,  mrib,  h96, BB, 256, 16, 96, 32, 1);
    k_lin<<<cdiv(BB * 16, TB), TB, 0, stream>>>(cog,  cogW,  cogb,  h96, BB,  64, 16, 96, 48, 1);
    k_lin<<<cdiv(BB * 16, TB), TB, 0, stream>>>(clin, clinW, clinb, h96, BB,  32, 16, 96, 64, 1);
    k_lin<<<cdiv(BB * 16, TB), TB, 0, stream>>>(gen,  genW,  genb,  h96, BB, 512, 16, 96, 80, 1);

    // ---- classifier ----
    k_lin<<<cdiv(BB * 64, TB), TB, 0, stream>>>(h96, c1W, c1b, t1, BB, 96, 64, 64, 0, 0);
    k_bn_small_stats<<<1, 64, 0, stream>>>(t1, cst, BB, 64);
    k_bn_small_apply_relu<<<cdiv(BB * 64, TB), TB, 0, stream>>>(t1, cst, cbn1_g, cbn1_b, BB, 64);

    k_lin<<<cdiv(BB * 32, TB), TB, 0, stream>>>(t1, c2W, c2b, t2, BB, 64, 32, 32, 0, 0);
    k_bn_small_stats<<<1, 32, 0, stream>>>(t2, cst, BB, 32);
    k_bn_small_apply_relu<<<cdiv(BB * 32, TB), TB, 0, stream>>>(t2, cst, cbn2_g, cbn2_b, BB, 32);

    k_cls3_logsoftmax<<<cdiv(BB, TB), TB, 0, stream>>>(t2, c3W, c3b, out, BB);
}
